// AttentionApproximator_14396730376896
// MI455X (gfx1250) — matrix operand order repaired and checked by reading, not compiled, not run on hardware
//
#include <hip/hip_runtime.h>
#include <hip/hip_bf16.h>

typedef __attribute__((ext_vector_type(16))) _Float16 v16h;
typedef __attribute__((ext_vector_type(8)))  _Float16 v8h;
typedef __attribute__((ext_vector_type(8)))  float    v8f;

#define B_  4
#define S_  2048
#define D_  16
#define K_  204   // max(1, int(2048*0.1))

__device__ __forceinline__ float relu_f32(float x) {
  // v_med3_f32 x, 0, +inf  -> single-instruction relu, no canonicalize
  return __builtin_amdgcn_fmed3f(x, 0.0f, __builtin_inff());
}

// ---------------- Phase A: scorer MLP + exact top-k (bitonic sort) ----------
__global__ __launch_bounds__(256) void score_topk_kernel(
    const float* __restrict__ full, const float* __restrict__ sw1,
    const float* __restrict__ sb1,  const float* __restrict__ sw2,
    const float* __restrict__ sb2,  int* __restrict__ top_idx)
{
  __shared__ float s_sw1[16 * 32];
  __shared__ float s_sb1[32];
  __shared__ float s_sw2[32];
  __shared__ float sc[S_];
  __shared__ int   si[S_];
  const int b = blockIdx.x;
  const int tid = threadIdx.x;

  for (int i = tid; i < 512; i += 256) s_sw1[i] = sw1[i];
  if (tid < 32) { s_sb1[tid] = sb1[tid]; s_sw2[tid] = sw2[tid]; }
  __syncthreads();

  const float b2 = sb2[0];
  for (int i = tid; i < S_; i += 256) {
    const float* fr = full + (size_t)(b * S_ + i) * D_;
    float r[16];
    #pragma unroll
    for (int t = 0; t < 16; t++) r[t] = fr[t];
    float s = b2;
    #pragma unroll
    for (int k = 0; k < 32; k++) {
      float h = s_sb1[k];
      #pragma unroll
      for (int t = 0; t < 16; t++) h += r[t] * s_sw1[t * 32 + k];
      h = relu_f32(h);
      s += h * s_sw2[k];
    }
    sc[i] = s; si[i] = i;
  }
  __syncthreads();

  // bitonic sort: descending by score, ascending index on ties (matches lax.top_k)
  for (int k = 2; k <= S_; k <<= 1) {
    for (int j = k >> 1; j > 0; j >>= 1) {
      for (int i = tid; i < S_; i += 256) {
        int ixj = i ^ j;
        if (ixj > i) {
          float av = sc[i], bv = sc[ixj];
          int   ai = si[i], bi = si[ixj];
          bool a_first = (av > bv) || (av == bv && ai < bi);
          bool desc = ((i & k) == 0);
          if (desc ? !a_first : a_first) {
            sc[i] = bv; sc[ixj] = av; si[i] = bi; si[ixj] = ai;
          }
        }
      }
      __syncthreads();
    }
  }
  for (int i = tid; i < K_; i += 256) top_idx[b * 256 + i] = si[i];
}

// ---------------- Phase B: WMMA main loop ----------
__global__ __launch_bounds__(256) void attn_wmma_kernel(
    const float* __restrict__ full, const float* __restrict__ mw1,
    const float* __restrict__ mb1,  const float* __restrict__ mw2,
    const float* __restrict__ mb2,  const float* __restrict__ mw3,
    const float* __restrict__ mb3,  const int* __restrict__ top_idx,
    float* __restrict__ out)
{
  __shared__ float    s_wq [16 * 64];   // w_q = mw1[0:16]
  __shared__ float    s_wkv[16 * 64];   // w_k + w_v
  __shared__ _Float16 s_kvb[K_ * 64];   // kv_j + mb1, packed f16 (row stride 128B)
  __shared__ float    s_tr [8][16 * 32];// per-wave D->A transpose buffer

  const int tid   = threadIdx.x;
  const int batch = blockIdx.x >> 4;    // 16 blocks per batch, uniform per block

  for (int i = tid; i < 1024; i += 256) {
    s_wq[i]  = mw1[i];
    s_wkv[i] = mw1[1024 + i] + mw1[2048 + i];
  }
  __syncthreads();

  // kv_j = full[top_idx[j]] @ (w_k+w_v) + mb1, stored f16
  for (int e = tid; e < K_ * 64; e += 256) {
    int j = e >> 6, kc = e & 63;
    int row = top_idx[batch * 256 + j];
    const float* fr = full + (size_t)(batch * S_ + row) * D_;
    float acc = mb1[kc];
    #pragma unroll
    for (int t = 0; t < 16; t++) acc += fr[t] * s_wkv[t * 64 + kc];
    s_kvb[e] = (_Float16)acc;
  }
  __syncthreads();

  const int  wave = tid >> 5, lane = tid & 31;
  const int  mrow = lane & 15;
  const bool hiL  = lane >= 16;
  const int  tile = blockIdx.x * 8 + wave;
  const int  s0   = (tile & 127) * 16;

  // A-fragment K offsets (16-bit A, 16x32): e<8 -> (hi?8:0)+e ; e>=8 -> 16+(hi?8:0)+(e-8)
  int akoff[16];
  #pragma unroll
  for (int e = 0; e < 16; e++) akoff[e] = (e < 8 ? e : 8 + e) + (hiL ? 8 : 0);

  // q A-fragments: q = full_row @ w_q, per-lane in A layout (f16)
  float fr[16];
  {
    const float* frp = full + (size_t)(batch * S_ + s0 + mrow) * D_;
    #pragma unroll
    for (int t = 0; t < 16; t++) fr[t] = frp[t];
  }
  v16h qA[2];
  #pragma unroll
  for (int c = 0; c < 2; c++) {
    #pragma unroll
    for (int e = 0; e < 16; e++) {
      int k = c * 32 + akoff[e];
      float q = 0.f;
      #pragma unroll
      for (int t = 0; t < 16; t++) q += fr[t] * s_wq[t * 64 + k];
      qA[c][e] = (_Float16)q;
    }
  }

  // B-fragments of mw2 (64x32): [col-tile n][K-half c]; B layout: lane=N, K=e<8 ? (hi?8:0)+e : 16+(hi?8:0)+(e-8)
  const int bcol = lane & 15;
  v16h Bh2[2][2];
  #pragma unroll
  for (int n = 0; n < 2; n++)
    #pragma unroll
    for (int c = 0; c < 2; c++)
      #pragma unroll
      for (int e = 0; e < 16; e++) {
        int k = c * 32 + (e < 8 ? e : 8 + e) + (hiL ? 8 : 0);
        Bh2[n][c][e] = (_Float16)mw2[k * 32 + n * 16 + bcol];
      }
  v16h Bm3;
  #pragma unroll
  for (int e = 0; e < 16; e++) {
    int k = (e < 8 ? e : 8 + e) + (hiL ? 8 : 0);
    Bm3[e] = (_Float16)mw3[k * 16 + bcol];
  }

  v8f cbias[2], H2acc[2];
  #pragma unroll
  for (int n = 0; n < 2; n++) {
    float bv = mb2[n * 16 + bcol];
    #pragma unroll
    for (int r = 0; r < 8; r++) { cbias[n][r] = bv; H2acc[n][r] = 0.f; }
  }

  // ---- main j loop: 204 iterations, 4 wmma each, bias via C operand ----
  const int kvlo = hiL ? 8 : 0;
  for (int j = 0; j < K_; j++) {
    const _Float16* kv = &s_kvb[j * 64];
    v16h a[2];
    #pragma unroll
    for (int c = 0; c < 2; c++) {
      v8h lo8 = *(const v8h*)(kv + c * 32 + kvlo);
      v8h hi8 = *(const v8h*)(kv + c * 32 + 16 + kvlo);
      v16h kvb = __builtin_shufflevector(lo8, hi8,
                   0,1,2,3,4,5,6,7,8,9,10,11,12,13,14,15);
      v16h t = kvb + qA[c];                       // q + kv + b1 (packed f16)
      #pragma unroll
      for (int e = 0; e < 16; e++)                // relu -> v_pk_max_num_f16
        t[e] = t[e] > (_Float16)0 ? t[e] : (_Float16)0;
      a[c] = t;
    }
    #pragma unroll
    for (int n = 0; n < 2; n++) {
      v8f d = __builtin_amdgcn_wmma_f32_16x16x32_f16(
                false, a[0], false, Bh2[n][0], (short)0, cbias[n], false, false);
      d = __builtin_amdgcn_wmma_f32_16x16x32_f16(
                false, a[1], false, Bh2[n][1], (short)0, d, false, false);
      #pragma unroll
      for (int r = 0; r < 8; r++)                 // relu + accumulate over j
        H2acc[n][r] += relu_f32(d[r]);            // v_med3_f32 + v_add_f32
    }
  }

  // ---- D-layout -> A-layout transpose through LDS (once per tile) ----
  {
    float* tb = s_tr[wave];
    int m0 = hiL ? 8 : 0;
    #pragma unroll
    for (int n = 0; n < 2; n++) {
      int col = n * 16 + (lane & 15);
      #pragma unroll
      for (int r = 0; r < 8; r++) tb[(m0 + r) * 32 + col] = H2acc[n][r];
    }
  }
  __syncthreads();
  v16h hA;
  {
    const float* tb = s_tr[wave];
    #pragma unroll
    for (int e = 0; e < 16; e++) hA[e] = (_Float16)tb[mrow * 32 + akoff[e]];
  }
  v8f zc = (v8f)0.0f;
  v8f dout = __builtin_amdgcn_wmma_f32_16x16x32_f16(
                false, hA, false, Bm3, (short)0, zc, false, false);

  // out = H2sum @ mw3 / 204 + mb3
  {
    const float inv = 1.0f / (float)K_;
    const float bo  = mb3[bcol];
    int m0 = hiL ? 8 : 0;
    #pragma unroll
    for (int r = 0; r < 8; r++) {
      int m = m0 + r;
      out[((size_t)(batch * S_) + s0 + m) * D_ + bcol] = dout[r] * inv + bo;
    }
  }
}

extern "C" void kernel_launch(void* const* d_in, const int* in_sizes, int n_in,
                              void* d_out, int out_size, void* d_ws, size_t ws_size,
                              hipStream_t stream) {
  const float* full = (const float*)d_in[0];
  const float* sw1  = (const float*)d_in[1];
  const float* sb1  = (const float*)d_in[2];
  const float* sw2  = (const float*)d_in[3];
  const float* sb2  = (const float*)d_in[4];
  const float* mw1  = (const float*)d_in[5];
  const float* mb1  = (const float*)d_in[6];
  const float* mw2  = (const float*)d_in[7];
  const float* mb2  = (const float*)d_in[8];
  const float* mw3  = (const float*)d_in[9];
  const float* mb3  = (const float*)d_in[10];
  int* top = (int*)d_ws;  // 4 * 256 ints

  score_topk_kernel<<<B_, 256, 0, stream>>>(full, sw1, sb1, sw2, sb2, top);
  attn_wmma_kernel<<<64, 256, 0, stream>>>(full, mw1, mb1, mw2, mb2, mw3, mb3,
                                           top, (float*)d_out);
}
